// CrossVariableAttention_54477365182962
// MI455X (gfx1250) — matrix operand order repaired and checked by reading, not compiled, not run on hardware
//
#include <hip/hip_runtime.h>
#include <hip/hip_bf16.h>

#define D_MODEL 512
#define NTOK    2048
#define NBATCH  8

typedef _Float16 v16h __attribute__((ext_vector_type(16)));
typedef _Float16 v8h  __attribute__((ext_vector_type(8)));
typedef float    v8f  __attribute__((ext_vector_type(8)));
typedef float    v4f  __attribute__((ext_vector_type(4)));

// ---------------- WMMA helpers (CDNA5 16x16x32 f16 -> f32) ----------------

__device__ __forceinline__ v8f wmma16(v16h a, v16h b, v8f c) {
    // 8 args: (neg_a, A, neg_b, B, c_mod, C, reuse_a, reuse_b)
    return __builtin_amdgcn_wmma_f32_16x16x32_f16(false, a, false, b, (short)0, c, false, false);
}

// A fragment (16x32, MxK): lane holds row m=lane&15; halfs [0..15] = K at
// koff..koff+15, koff = (lane>>4)*16.
__device__ __forceinline__ v16h load_afrag(const _Float16* p) {
    v8h lo = *(const v8h*)p;
    v8h hi = *(const v8h*)(p + 8);
    v16h r;
#pragma unroll
    for (int i = 0; i < 8; ++i) { r[i] = lo[i]; r[i + 8] = hi[i]; }
    return r;
}

// B fragment (32x16, KxN), fed from Bt[n][k] row-major: lane holds column
// n=lane&15; 16 contiguous halfs starting at k = (lane>>4)*16.
__device__ __forceinline__ v16h load_bfrag(const _Float16* p) {
    return *(const v16h*)p;
}

// ---------------- transpose + f32->f16 convert ----------------
// src: [R][C] f32 (batched via blockIdx.z), dst: [C][R] f16
__global__ __launch_bounds__(256) void k_transpose_cvt(const float* __restrict__ src,
                                                       _Float16* __restrict__ dst,
                                                       int R, int C) {
    __shared__ float tile[32][33];
    const size_t bofs = (size_t)blockIdx.z * R * C;
    src += bofs; dst += bofs;
    const int c0 = blockIdx.x * 32, r0 = blockIdx.y * 32;
    const int tx = threadIdx.x, ty = threadIdx.y;
#pragma unroll
    for (int i = 0; i < 4; ++i)
        tile[ty + i * 8][tx] = src[(size_t)(r0 + ty + i * 8) * C + c0 + tx];
    __syncthreads();
#pragma unroll
    for (int i = 0; i < 4; ++i)
        dst[(size_t)(c0 + ty + i * 8) * R + r0 + tx] = (_Float16)tile[tx][ty + i * 8];
}

// ---------------- QKV GEMM: qkv = xh @ Wqkv^T^T + b ----------------
// A = xh [N][512] f16, Bt = wqkvt [1536][512] f16. Wave -> 16x64 tile.
// j<512 -> q row-major; j<1024 -> k row-major; else v stored transposed [d][n].
__global__ __launch_bounds__(256) void k_qkv_gemm(const _Float16* __restrict__ xh,
                                                  const _Float16* __restrict__ wt,
                                                  const float* __restrict__ bqkv,
                                                  _Float16* __restrict__ q,
                                                  _Float16* __restrict__ kmat,
                                                  _Float16* __restrict__ vt) {
    const int lane = threadIdx.x & 31, wave = threadIdx.x >> 5;
    const int b = blockIdx.z;
    const int mbase = blockIdx.x * 128 + wave * 16;
    const int nbase = blockIdx.y * 64;
    const int koff = (lane >> 4) * 16, bsel = (lane >> 4) * 16;

    const _Float16* Ar  = xh + ((size_t)b * NTOK + mbase + (lane & 15)) * D_MODEL;
    const _Float16* Br0 = wt + (size_t)(nbase + (lane & 15)) * D_MODEL;

    v8f acc[4] = {v8f{}, v8f{}, v8f{}, v8f{}};
#pragma unroll 2
    for (int kk = 0; kk < D_MODEL; kk += 32) {
        v16h a = load_afrag(Ar + kk + koff);
#pragma unroll
        for (int t = 0; t < 4; ++t) {
            v16h bb = load_bfrag(Br0 + (size_t)t * 16 * D_MODEL + kk + bsel);
            acc[t] = wmma16(a, bb, acc[t]);
        }
    }

    const int m0 = mbase + 8 * (lane >> 4);
#pragma unroll
    for (int t = 0; t < 4; ++t) {
        const int j = nbase + t * 16 + (lane & 15);
        const float bv = bqkv[j];
        if (j < 512) {
            _Float16* d = q + (size_t)b * NTOK * D_MODEL;
#pragma unroll
            for (int r = 0; r < 8; ++r)
                d[(size_t)(m0 + r) * D_MODEL + j] = (_Float16)(acc[t][r] + bv);
        } else if (j < 1024) {
            _Float16* d = kmat + (size_t)b * NTOK * D_MODEL;
#pragma unroll
            for (int r = 0; r < 8; ++r)
                d[(size_t)(m0 + r) * D_MODEL + (j - 512)] = (_Float16)(acc[t][r] + bv);
        } else {
            _Float16* d = vt + (size_t)b * D_MODEL * NTOK + (size_t)(j - 1024) * NTOK + m0;
            v8h hv;
#pragma unroll
            for (int r = 0; r < 8; ++r) hv[r] = (_Float16)(acc[t][r] + bv);
            *(v8h*)d = hv;
        }
    }
}

// ---------------- fused scores + softmax ----------------
// Block: 16 rows x 2048 cols. Scores via WMMA into 64KB LDS (f16), then
// per-row wave32 shuffle reductions, write normalized probs f16.
__global__ __launch_bounds__(256) void k_attn_softmax(const _Float16* __restrict__ q,
                                                      const _Float16* __restrict__ kmat,
                                                      _Float16* __restrict__ probs) {
    __shared__ _Float16 sc[16 * NTOK];   // 64 KB
    const int lane = threadIdx.x & 31, wave = threadIdx.x >> 5;
    const int b = blockIdx.y, rbase = blockIdx.x * 16;
    const float scale = 0.04419417382415922f;  // 1/sqrt(512)
    const int koff = (lane >> 4) * 16, bsel = (lane >> 4) * 16;

    const _Float16* Ar = q + ((size_t)b * NTOK + rbase + (lane & 15)) * D_MODEL;
    const _Float16* Kb = kmat + (size_t)b * NTOK * D_MODEL;

    for (int ct = wave; ct < NTOK / 16; ct += 8) {
        const _Float16* Br = Kb + (size_t)(ct * 16 + (lane & 15)) * D_MODEL;
        v8f acc = {};
#pragma unroll 2
        for (int kk = 0; kk < D_MODEL; kk += 32) {
            v16h a  = load_afrag(Ar + kk + koff);
            v16h bb = load_bfrag(Br + kk + bsel);
            acc = wmma16(a, bb, acc);
        }
        const int col = ct * 16 + (lane & 15);
        const int m0 = 8 * (lane >> 4);
#pragma unroll
        for (int r = 0; r < 8; ++r)
            sc[(m0 + r) * NTOK + col] = (_Float16)(acc[r] * scale);
    }
    __syncthreads();

    for (int rr = wave; rr < 16; rr += 8) {
        const _Float16* row = sc + rr * NTOK;
        float mx = -3.0e38f;
        for (int i = lane; i < NTOK; i += 32) mx = fmaxf(mx, (float)row[i]);
#pragma unroll
        for (int off = 16; off > 0; off >>= 1) mx = fmaxf(mx, __shfl_xor(mx, off, 32));
        float s = 0.f;
        for (int i = lane; i < NTOK; i += 32) s += __expf((float)row[i] - mx);
#pragma unroll
        for (int off = 16; off > 0; off >>= 1) s += __shfl_xor(s, off, 32);
        const float inv = 1.f / s;
        _Float16* pr = probs + ((size_t)b * NTOK + rbase + rr) * NTOK;
        for (int i = lane; i < NTOK; i += 32)
            pr[i] = (_Float16)(__expf((float)row[i] - mx) * inv);
    }
}

// ---------------- P @ V : ctx[n][d] = probs[n][:] . vt[d][:] ----------------
__global__ __launch_bounds__(256) void k_pv_gemm(const _Float16* __restrict__ probs,
                                                 const _Float16* __restrict__ vt,
                                                 _Float16* __restrict__ ctx) {
    const int lane = threadIdx.x & 31, wave = threadIdx.x >> 5;
    const int b = blockIdx.z;
    const int mbase = blockIdx.x * 128 + wave * 16;
    const int nbase = blockIdx.y * 64;
    const int koff = (lane >> 4) * 16, bsel = (lane >> 4) * 16;

    const _Float16* Ar  = probs + ((size_t)b * NTOK + mbase + (lane & 15)) * NTOK;
    const _Float16* Br0 = vt + (size_t)b * D_MODEL * NTOK + (size_t)(nbase + (lane & 15)) * NTOK;

    v8f acc[4] = {v8f{}, v8f{}, v8f{}, v8f{}};
#pragma unroll 2
    for (int kk = 0; kk < NTOK; kk += 32) {
        v16h a = load_afrag(Ar + kk + koff);
#pragma unroll
        for (int t = 0; t < 4; ++t) {
            v16h bb = load_bfrag(Br0 + (size_t)t * 16 * NTOK + kk + bsel);
            acc[t] = wmma16(a, bb, acc[t]);
        }
    }
    const int m0 = mbase + 8 * (lane >> 4);
    _Float16* d = ctx + (size_t)b * NTOK * D_MODEL;
#pragma unroll
    for (int t = 0; t < 4; ++t) {
        const int j = nbase + t * 16 + (lane & 15);
#pragma unroll
        for (int r = 0; r < 8; ++r)
            d[(size_t)(m0 + r) * D_MODEL + j] = (_Float16)acc[t][r];
    }
}

// ---------------- proj + bias, store transposed f32 out[b][e][n] ----------------
__global__ __launch_bounds__(256) void k_proj_gemm(const _Float16* __restrict__ ctx,
                                                   const _Float16* __restrict__ wpt,
                                                   const float* __restrict__ bproj,
                                                   float* __restrict__ out) {
    const int lane = threadIdx.x & 31, wave = threadIdx.x >> 5;
    const int b = blockIdx.z;
    const int mbase = blockIdx.x * 128 + wave * 16;
    const int nbase = blockIdx.y * 64;
    const int koff = (lane >> 4) * 16, bsel = (lane >> 4) * 16;

    const _Float16* Ar  = ctx + ((size_t)b * NTOK + mbase + (lane & 15)) * D_MODEL;
    const _Float16* Br0 = wpt + (size_t)(nbase + (lane & 15)) * D_MODEL;

    v8f acc[4] = {v8f{}, v8f{}, v8f{}, v8f{}};
#pragma unroll 2
    for (int kk = 0; kk < D_MODEL; kk += 32) {
        v16h a = load_afrag(Ar + kk + koff);
#pragma unroll
        for (int t = 0; t < 4; ++t) {
            v16h bb = load_bfrag(Br0 + (size_t)t * 16 * D_MODEL + kk + bsel);
            acc[t] = wmma16(a, bb, acc[t]);
        }
    }
    const int m0 = mbase + 8 * (lane >> 4);
#pragma unroll
    for (int t = 0; t < 4; ++t) {
        const int j = nbase + t * 16 + (lane & 15);   // output feature e
        const float bv = bproj[j];
        float* d = out + (size_t)b * D_MODEL * NTOK + (size_t)j * NTOK + m0;
        v4f lo, hi;
#pragma unroll
        for (int r = 0; r < 4; ++r) { lo[r] = acc[t][r] + bv; hi[r] = acc[t][r + 4] + bv; }
        *(v4f*)d = lo;
        *(v4f*)(d + 4) = hi;
    }
}

// ---------------- launcher ----------------
extern "C" void kernel_launch(void* const* d_in, const int* in_sizes, int n_in,
                              void* d_out, int out_size, void* d_ws, size_t ws_size,
                              hipStream_t stream) {
    const float* x     = (const float*)d_in[0];   // [8][512][2048]
    const float* Wqkv  = (const float*)d_in[1];   // [512][1536]
    const float* bqkv  = (const float*)d_in[2];   // [1536]
    const float* Wproj = (const float*)d_in[3];   // [512][512]
    const float* bproj = (const float*)d_in[4];   // [512]
    float* out = (float*)d_out;                   // [8][512][2048]

    char* ws = (char*)d_ws;
    size_t off = 0;
    auto carve = [&](size_t bytes) -> void* {
        void* p = ws + off;
        off += (bytes + 255) & ~(size_t)255;
        return p;
    };
    _Float16* xh    = (_Float16*)carve((size_t)NBATCH * NTOK * D_MODEL * 2); // x^T f16
    _Float16* wqkvt = (_Float16*)carve((size_t)1536 * D_MODEL * 2);          // Wqkv^T f16
    _Float16* wpt   = (_Float16*)carve((size_t)D_MODEL * D_MODEL * 2);       // Wproj^T f16
    _Float16* q     = (_Float16*)carve((size_t)NBATCH * NTOK * D_MODEL * 2);
    _Float16* kbuf  = (_Float16*)carve((size_t)NBATCH * NTOK * D_MODEL * 2);
    _Float16* vt    = (_Float16*)carve((size_t)NBATCH * D_MODEL * NTOK * 2); // V^T
    _Float16* probs = (_Float16*)carve((size_t)NBATCH * NTOK * NTOK * 2);
    _Float16* ctx   = (_Float16*)carve((size_t)NBATCH * NTOK * D_MODEL * 2);

    dim3 blkT(32, 8);
    // x [512][2048] -> xh [2048][512], per batch
    k_transpose_cvt<<<dim3(NTOK / 32, D_MODEL / 32, NBATCH), blkT, 0, stream>>>(x, xh, D_MODEL, NTOK);
    // Wqkv [512][1536] -> wqkvt [1536][512]
    k_transpose_cvt<<<dim3(1536 / 32, D_MODEL / 32, 1), blkT, 0, stream>>>(Wqkv, wqkvt, D_MODEL, 1536);
    // Wproj [512][512] -> wpt [512][512]
    k_transpose_cvt<<<dim3(D_MODEL / 32, D_MODEL / 32, 1), blkT, 0, stream>>>(Wproj, wpt, D_MODEL, D_MODEL);

    k_qkv_gemm<<<dim3(NTOK / 128, 1536 / 64, NBATCH), 256, 0, stream>>>(xh, wqkvt, bqkv, q, kbuf, vt);
    k_attn_softmax<<<dim3(NTOK / 16, NBATCH), 256, 0, stream>>>(q, kbuf, probs);
    k_pv_gemm<<<dim3(NTOK / 128, D_MODEL / 64, NBATCH), 256, 0, stream>>>(probs, vt, ctx);
    k_proj_gemm<<<dim3(NTOK / 128, D_MODEL / 64, NBATCH), 256, 0, stream>>>(ctx, wpt, bproj, out);
}
